// HGNNEncoder_85074712199280
// MI455X (gfx1250) — matrix operand order repaired and checked by reading, not compiled, not run on hardware
//
#include <hip/hip_runtime.h>

typedef __attribute__((ext_vector_type(16))) _Float16 v16h;
typedef __attribute__((ext_vector_type(8)))  _Float16 v8h;
typedef __attribute__((ext_vector_type(8)))  float    v8f;

#define N_NODES 100000
#define N_EDGES 100000
#define NNZ     1600000
#define TB      256

// ---------------------------------------------------------------- utilities
__global__ void zero_f32(float* __restrict__ p, int n) {
    int i = blockIdx.x * blockDim.x + threadIdx.x;
    if (i < n) p[i] = 0.0f;
}

__global__ void f32_to_f16(const float* __restrict__ s, _Float16* __restrict__ d, int n) {
    int i = blockIdx.x * blockDim.x + threadIdx.x;
    if (i < n) d[i] = (_Float16)s[i];
}

// W[K][N] row-major -> Wt[N][K] fp16 (columns become contiguous rows)
__global__ void conv_transpose_w(const float* __restrict__ W, _Float16* __restrict__ Wt,
                                 int K, int N) {
    int i = blockIdx.x * blockDim.x + threadIdx.x;
    if (i < K * N) {
        int k = i / N, n = i % N;
        Wt[(size_t)n * K + k] = (_Float16)W[i];
    }
}

// ---------------------------------------------------------------- degrees
__global__ void degrees_kernel(const long long* __restrict__ nidx,
                               const long long* __restrict__ eidx,
                               float* __restrict__ dv, float* __restrict__ de, int nnz) {
    int i = blockIdx.x * blockDim.x + threadIdx.x;
    if (i < nnz) {
        atomicAdd(&dv[(int)nidx[i]], 1.0f);
        atomicAdd(&de[(int)eidx[i]], 1.0f);
    }
}

// ---------------------------------------------------------------- WMMA GEMM
// C[M,N] = A16[M,128] * Bt16[N,128]^T   (fp16 in, fp32 accumulate)
// 256-thread block = 8 waves; wave -> one 16x16 C tile.
// colTiles = N/16 must divide 8; block covers (8/colTiles)*16 rows.
__global__ void gemm_wmma_f16(const _Float16* __restrict__ A,
                              const _Float16* __restrict__ Bt,
                              float* __restrict__ C,
                              int M, int N, int colTiles) {
    const int wave = threadIdx.x >> 5;
    const int lane = threadIdx.x & 31;
    const int half = lane >> 4;     // lane-half selector
    const int l15  = lane & 15;

    const int rowTiles = 8 / colTiles;
    const int tileR = wave / colTiles;
    const int tileC = wave % colTiles;
    const int row_base = blockIdx.x * (rowTiles * 16) + tileR * 16;
    const int col_base = tileC * 16;
    if (row_base >= M) return;

    // A fragment (16x32 f16): halves 0-7 = K(kb + 16*half + j), halves 8-15 = K(kb+16*half+8+j)
    const _Float16* arow = A + (size_t)(row_base + l15) * 128;
    // B fragment (32x16 f16): lane-half splits K into 16-blocks, contiguous within
    const _Float16* brow = Bt + (size_t)(col_base + l15) * 128 + half * 16;

    v8f c = {};
#pragma unroll
    for (int kb = 0; kb < 128; kb += 32) {
        v8h a0 = *(const v8h*)(arow + kb + half * 16);
        v8h a1 = *(const v8h*)(arow + kb + half * 16 + 8);
        v16h a, b;
        b = *(const v16h*)(brow + kb);
#pragma unroll
        for (int j = 0; j < 8; ++j) { a[j] = a0[j]; a[j + 8] = a1[j]; }
        c = __builtin_amdgcn_wmma_f32_16x16x32_f16(
                /*neg_a=*/false, a, /*neg_b=*/false, b,
                /*c_mod=*/(short)0, c, /*reuse_a=*/false, /*reuse_b=*/false);
    }

    // D layout: VGPR r -> row (r + 8*half), lane l15 -> col
    float* crow = C + (size_t)(row_base + half * 8) * N + col_base + l15;
#pragma unroll
    for (int r = 0; r < 8; ++r) crow[(size_t)r * N] = c[r];
}

// ---------------------------------------------------------------- message passing
// For each nz, add src[sidx[nz], :] into dst[didx[nz], :]; 4 floats per thread.
__global__ void scatter_add4(const float* __restrict__ src,
                             const long long* __restrict__ sidx,
                             const long long* __restrict__ didx,
                             float* __restrict__ dst,
                             int nnz, int F, int shift) {
    long long t = (long long)blockIdx.x * blockDim.x + threadIdx.x;
    int nz = (int)(t >> shift);
    if (nz >= nnz) return;
    int c = ((int)t & ((1 << shift) - 1)) * 4;
    int s = (int)sidx[nz];
    int d = (int)didx[nz];
    const float4 v = *reinterpret_cast<const float4*>(src + (size_t)s * F + c);
    float* dp = dst + (size_t)d * F + c;
    atomicAdd(dp + 0, v.x);
    atomicAdd(dp + 1, v.y);
    atomicAdd(dp + 2, v.z);
    atomicAdd(dp + 3, v.w);
}

// m[r, :] *= (deg[r] > 0 ? 1/deg[r] : 0)
__global__ void scale_invdeg(float* __restrict__ m, const float* __restrict__ deg,
                             int rows, int F) {
    int i = blockIdx.x * blockDim.x + threadIdx.x;
    if (i < rows * F) {
        int r = i / F;
        float dg = deg[r];
        m[i] *= (dg > 0.0f) ? (1.0f / dg) : 0.0f;
    }
}

// h16 = fp16(relu(acc * d_inv + bias))
__global__ void finalize_relu_f16(const float* __restrict__ acc, const float* __restrict__ deg,
                                  const float* __restrict__ bias, _Float16* __restrict__ h16,
                                  int rows, int F) {
    int i = blockIdx.x * blockDim.x + threadIdx.x;
    if (i < rows * F) {
        int r = i / F, c = i % F;
        float dg = deg[r];
        float inv = (dg > 0.0f) ? (1.0f / dg) : 0.0f;
        float v = acc[i] * inv + bias[c];
        h16[i] = (_Float16)(v > 0.0f ? v : 0.0f);
    }
}

// out = out * d_inv + bias   (in place on d_out)
__global__ void finalize_out(float* __restrict__ out, const float* __restrict__ deg,
                             const float* __restrict__ bias, int rows, int F) {
    int i = blockIdx.x * blockDim.x + threadIdx.x;
    if (i < rows * F) {
        int r = i / F, c = i % F;
        float dg = deg[r];
        float inv = (dg > 0.0f) ? (1.0f / dg) : 0.0f;
        out[i] = out[i] * inv + bias[c];
    }
}

// ---------------------------------------------------------------- launch
extern "C" void kernel_launch(void* const* d_in, const int* in_sizes, int n_in,
                              void* d_out, int out_size, void* d_ws, size_t ws_size,
                              hipStream_t stream) {
    const float*     x    = (const float*)d_in[0];
    const long long* hidx = (const long long*)d_in[1];   // int64 [2, NNZ]
    const float*     W1   = (const float*)d_in[2];
    const float*     b1   = (const float*)d_in[3];
    const float*     W2   = (const float*)d_in[4];
    const float*     b2   = (const float*)d_in[5];
    float*           out  = (float*)d_out;

    const long long* nidx = hidx;         // row 0: node indices
    const long long* eidx = hidx + NNZ;   // row 1: hyperedge indices

    // workspace carve-out (256B aligned)
    char* w = (char*)d_ws;
    auto carve = [&](size_t bytes) -> char* {
        char* p = w;
        w += (bytes + 255) & ~(size_t)255;
        return p;
    };
    float*    deg_v = (float*)carve((size_t)N_NODES * 4);
    float*    deg_e = (float*)carve((size_t)N_EDGES * 4);
    _Float16* x16   = (_Float16*)carve((size_t)N_NODES * 128 * 2);  // reused as h16
    _Float16* W1t   = (_Float16*)carve(128 * 128 * 2);
    _Float16* W2t   = (_Float16*)carve(64 * 128 * 2);
    float*    xw    = (float*)carve((size_t)N_NODES * 128 * 4);     // xw1, then xw2
    float*    me    = (float*)carve((size_t)N_EDGES * 128 * 4);     // me1, then me2
    float*    acc   = (float*)carve((size_t)N_NODES * 128 * 4);     // layer-1 node accumulator
    (void)ws_size; (void)in_sizes; (void)n_in; (void)out_size;

    auto nb = [](long long n) { return (unsigned)((n + TB - 1) / TB); };

    // ---- zero all accumulators (fresh every launch: determinism) ----
    zero_f32<<<nb(N_NODES), TB, 0, stream>>>(deg_v, N_NODES);
    zero_f32<<<nb(N_EDGES), TB, 0, stream>>>(deg_e, N_EDGES);
    zero_f32<<<nb((long long)N_EDGES * 128), TB, 0, stream>>>(me, N_EDGES * 128);
    zero_f32<<<nb((long long)N_NODES * 128), TB, 0, stream>>>(acc, N_NODES * 128);

    // ---- degrees + fp16 staging ----
    degrees_kernel<<<nb(NNZ), TB, 0, stream>>>(nidx, eidx, deg_v, deg_e, NNZ);
    f32_to_f16<<<nb((long long)N_NODES * 128), TB, 0, stream>>>(x, x16, N_NODES * 128);
    conv_transpose_w<<<nb(128 * 128), TB, 0, stream>>>(W1, W1t, 128, 128);
    conv_transpose_w<<<nb(128 * 64), TB, 0, stream>>>(W2, W2t, 128, 64);

    // ================= layer 1 =================
    // xw1 = x @ W1  (WMMA)
    gemm_wmma_f16<<<N_NODES / 16, TB, 0, stream>>>(x16, W1t, xw, N_NODES, 128, 8);
    // node -> hyperedge
    scatter_add4<<<nb((long long)NNZ * 32), TB, 0, stream>>>(xw, nidx, eidx, me, NNZ, 128, 5);
    scale_invdeg<<<nb((long long)N_EDGES * 128), TB, 0, stream>>>(me, deg_e, N_EDGES, 128);
    // hyperedge -> node
    scatter_add4<<<nb((long long)NNZ * 32), TB, 0, stream>>>(me, eidx, nidx, acc, NNZ, 128, 5);
    // h = relu(acc * d_inv + b1), emitted directly as fp16 for GEMM2 (reuses x16)
    finalize_relu_f16<<<nb((long long)N_NODES * 128), TB, 0, stream>>>(
        acc, deg_v, b1, x16, N_NODES, 128);

    // ================= layer 2 =================
    zero_f32<<<nb((long long)N_EDGES * 64), TB, 0, stream>>>(me, N_EDGES * 64);
    zero_f32<<<nb((long long)N_NODES * 64), TB, 0, stream>>>(out, N_NODES * 64);
    // xw2 = h @ W2  (WMMA) — 8 waves = 2 row-tiles x 4 col-tiles
    gemm_wmma_f16<<<N_NODES / 32, TB, 0, stream>>>(x16, W2t, xw, N_NODES, 64, 4);
    // node -> hyperedge
    scatter_add4<<<nb((long long)NNZ * 16), TB, 0, stream>>>(xw, nidx, eidx, me, NNZ, 64, 4);
    scale_invdeg<<<nb((long long)N_EDGES * 64), TB, 0, stream>>>(me, deg_e, N_EDGES, 64);
    // hyperedge -> node, straight into d_out
    scatter_add4<<<nb((long long)NNZ * 16), TB, 0, stream>>>(me, eidx, nidx, out, NNZ, 64, 4);
    finalize_out<<<nb((long long)N_NODES * 64), TB, 0, stream>>>(out, deg_v, b2, N_NODES, 64);
}
